// sLSTMCell_15461882265859
// MI455X (gfx1250) — matrix operand order repaired and checked by reading, not compiled, not run on hardware
//
#include <hip/hip_runtime.h>
#include <hip/hip_bf16.h>

typedef __attribute__((ext_vector_type(16))) __bf16 v16bf;
typedef __attribute__((ext_vector_type(8)))  float  v8f;

union FragU { v16bf v; uint4 q[2]; };

// ---------- fp32 <-> bf16 helpers (round-to-nearest-even) ----------
__device__ __forceinline__ unsigned short f2bf(float f) {
  unsigned int u = __float_as_uint(f);
  u += 0x7FFFu + ((u >> 16) & 1u);
  return (unsigned short)(u >> 16);
}
__device__ __forceinline__ float bf2f(unsigned short h) {
  return __uint_as_float(((unsigned int)h) << 16);
}

// Convert 4 fp32 to (hi, lo) bf16 pairs and store packed to LDS (8-byte aligned).
__device__ __forceinline__ void cvt_store4(unsigned short* hi, unsigned short* lo, float4 v) {
  unsigned short h0 = f2bf(v.x), h1 = f2bf(v.y), h2 = f2bf(v.z), h3 = f2bf(v.w);
  unsigned short l0 = f2bf(v.x - bf2f(h0));
  unsigned short l1 = f2bf(v.y - bf2f(h1));
  unsigned short l2 = f2bf(v.z - bf2f(h2));
  unsigned short l3 = f2bf(v.w - bf2f(h3));
  uint2 ph, pl;
  ph.x = (unsigned)h0 | ((unsigned)h1 << 16);
  ph.y = (unsigned)h2 | ((unsigned)h3 << 16);
  pl.x = (unsigned)l0 | ((unsigned)l1 << 16);
  pl.y = (unsigned)l2 | ((unsigned)l3 << 16);
  *(uint2*)hi = ph;
  *(uint2*)lo = pl;
}

// ---------- WMMA fragment assembly from LDS tiles (row-major, 32 halfwords/row) ----------
// A (16x32, 16-bit): lanes 0-15 = M, half=lane>>4:
//   elements 0..7  -> K = 8*half + e        (uint4 index `half`)
//   elements 8..15 -> K = 16 + 8*half + e-8 (uint4 index `2+half`)
__device__ __forceinline__ v16bf load_frag_A(const unsigned short* base, int row, int half) {
  FragU f;
  const uint4* p = (const uint4*)(base + row * 32);
  f.q[0] = p[half];
  f.q[1] = p[2 + half];
  return f.v;
}
// B (32x16, 16-bit): lanes 0-15 = N, half=lane>>4: same K order as A
__device__ __forceinline__ v16bf load_frag_B(const unsigned short* base, int row, int half) {
  FragU f;
  const uint4* p = (const uint4*)(base + row * 32);
  f.q[0] = p[half];
  f.q[1] = p[2 + half];
  return f.v;
}

// ---------- GEMM: C[M,N] = A[M,K] * W[N,K]^T  via 3x-bf16 split WMMA ----------
// Workgroup tile 128x128, 8 waves (4 M-slices x 2 N-slices), K-step 32.
__global__ __launch_bounds__(256) void gemm_nt_bf16x3(
    const float* __restrict__ A, const float* __restrict__ W,
    float* __restrict__ C, int N, int K)
{
  __shared__ unsigned short lAhi[128 * 32];
  __shared__ unsigned short lAlo[128 * 32];
  __shared__ unsigned short lBhi[128 * 32];
  __shared__ unsigned short lBlo[128 * 32];

  const int tid  = threadIdx.x;
  const int row0 = blockIdx.x * 128;
  const int col0 = blockIdx.y * 128;
  const int w    = tid >> 5;
  const int lane = tid & 31;
  const int half = lane >> 4;
  const int ln16 = lane & 15;
  const int wm = w & 3;   // 0..3 -> M slice of 32
  const int wn = w >> 2;  // 0..1 -> N slice of 64

  v8f acc[2][4];
#pragma unroll
  for (int a = 0; a < 2; ++a)
#pragma unroll
    for (int b = 0; b < 4; ++b)
#pragma unroll
      for (int e = 0; e < 8; ++e) acc[a][b][e] = 0.0f;

  for (int kb = 0; kb < K; kb += 32) {
    // Stage 128x32 tiles of A and W as (hi, lo) bf16 in LDS.
#pragma unroll
    for (int j = 0; j < 4; ++j) {
      int qi = tid + 256 * j;          // 0..1023 quads (128 rows x 8 quads)
      int r  = qi >> 3;
      int c4 = (qi & 7) << 2;
      float4 xa = *(const float4*)(A + (size_t)(row0 + r) * K + kb + c4);
      float4 xb = *(const float4*)(W + (size_t)(col0 + r) * K + kb + c4);
      int o = r * 32 + c4;
      cvt_store4(lAhi + o, lAlo + o, xa);
      cvt_store4(lBhi + o, lBlo + o, xb);
    }
    __syncthreads();

    v16bf bh[4], bl[4];
#pragma unroll
    for (int nt = 0; nt < 4; ++nt) {
      int n = wn * 64 + nt * 16 + ln16;
      bh[nt] = load_frag_B(lBhi, n, half);
      bl[nt] = load_frag_B(lBlo, n, half);
    }
#pragma unroll
    for (int mt = 0; mt < 2; ++mt) {
      int m = wm * 32 + mt * 16 + ln16;
      v16bf ah = load_frag_A(lAhi, m, half);
      v16bf al = load_frag_A(lAlo, m, half);
#pragma unroll
      for (int nt = 0; nt < 4; ++nt) {
        acc[mt][nt] = __builtin_amdgcn_wmma_f32_16x16x32_bf16(
            false, ah, false, bh[nt], (short)0, acc[mt][nt], false, false);
        acc[mt][nt] = __builtin_amdgcn_wmma_f32_16x16x32_bf16(
            false, ah, false, bl[nt], (short)0, acc[mt][nt], false, false);
        acc[mt][nt] = __builtin_amdgcn_wmma_f32_16x16x32_bf16(
            false, al, false, bh[nt], (short)0, acc[mt][nt], false, false);
      }
    }
    __syncthreads();
  }

  // Epilogue: C/D layout -> lane n = ln16, VGPR r -> m = r + 8*half.
#pragma unroll
  for (int mt = 0; mt < 2; ++mt) {
#pragma unroll
    for (int nt = 0; nt < 4; ++nt) {
      int gm = row0 + wm * 32 + mt * 16 + 8 * half;
      int gn = col0 + wn * 64 + nt * 16 + ln16;
#pragma unroll
      for (int r = 0; r < 8; ++r) {
        C[(size_t)(gm + r) * N + gn] = acc[mt][nt][r];
      }
    }
  }
}

// ---------- block-wide sum over 256 threads (wave32) ----------
__device__ __forceinline__ float blockSum(float v, float* sbuf) {
#pragma unroll
  for (int o = 16; o > 0; o >>= 1) v += __shfl_down(v, o, 32);
  const int lane = threadIdx.x & 31;
  const int wid  = threadIdx.x >> 5;
  if (lane == 0) sbuf[wid] = v;
  __syncthreads();
  if (wid == 0) {
    float x = (lane < 8) ? sbuf[lane] : 0.0f;
#pragma unroll
    for (int o = 4; o > 0; o >>= 1) x += __shfl_down(x, o, 32);
    if (lane == 0) sbuf[8] = x;
  }
  __syncthreads();
  float r = sbuf[8];
  __syncthreads();
  return r;
}

// ---------- fused per-row sLSTM pointwise stage ----------
// One 256-thread block per batch row; each thread owns 16 of 4096 gate
// columns: column j = t + 256*s; gate chunk = s/4, in-gate pos p = t + 256*(s%4).
__global__ __launch_bounds__(256) void slstm_pointwise(
    const float* __restrict__ gx, const float* __restrict__ gh,
    const float* __restrict__ c_prev,
    const float* __restrict__ bx, const float* __restrict__ bh,
    const float* __restrict__ ln_in_g, const float* __restrict__ ln_in_b,
    const float* __restrict__ ln_hu_g, const float* __restrict__ ln_hu_b,
    const float* __restrict__ ln_c_g, const float* __restrict__ ln_c_b,
    const float* __restrict__ ln_h_g, const float* __restrict__ ln_h_b,
    float* __restrict__ h_out, float* __restrict__ c_out, int H)
{
  __shared__ float sred[16];
  const int row = blockIdx.x;
  const int t = threadIdx.x;
  const int FH = 4 * H;
  const float invFH = 1.0f / (float)FH;
  const float invH  = 1.0f / (float)H;
  const float* gxr = gx + (size_t)row * FH;
  const float* ghr = gh + (size_t)row * FH;

  float gxv[16], ghv[16];
  float sx = 0.0f, sh = 0.0f;
#pragma unroll
  for (int s = 0; s < 16; ++s) {
    int j = t + 256 * s;
    gxv[s] = gxr[j] + bx[j];
    ghv[s] = ghr[j] + bh[j];
    sx += gxv[s];
    sh += ghv[s];
  }
  float mux = blockSum(sx, sred) * invFH;
  float muh = blockSum(sh, sred) * invFH;

  float vx = 0.0f, vh = 0.0f;
#pragma unroll
  for (int s = 0; s < 16; ++s) {
    float d0 = gxv[s] - mux; vx += d0 * d0;
    float d1 = ghv[s] - muh; vh += d1 * d1;
  }
  vx = blockSum(vx, sred) * invFH;
  vh = blockSum(vh, sred) * invFH;
  float rsx = rsqrtf(vx + 1e-5f);
  float rsh = rsqrtf(vh + 1e-5f);

  float gates[16];
#pragma unroll
  for (int s = 0; s < 16; ++s) {
    int j = t + 256 * s;
    gates[s] = (gxv[s] - mux) * rsx * ln_in_g[j] + ln_in_b[j]
             + (ghv[s] - muh) * rsh * ln_hu_g[j] + ln_hu_b[j];
  }

  // normalized_exp_gate for i (s=0..3) and f (s=4..7)
  float si = gates[0] + gates[1] + gates[2] + gates[3];
  float mi = blockSum(si, sred) * invH;
  float ei[4]; float sei = 0.0f;
#pragma unroll
  for (int q = 0; q < 4; ++q) {
    float cg = fminf(5.0f, fmaxf(-5.0f, gates[q] - mi));
    ei[q] = expf(cg);
    sei += ei[q];
  }
  sei = blockSum(sei, sred);

  float sf = gates[4] + gates[5] + gates[6] + gates[7];
  float mf = blockSum(sf, sred) * invH;
  float ef[4]; float sef = 0.0f;
#pragma unroll
  for (int q = 0; q < 4; ++q) {
    float cg = fminf(5.0f, fmaxf(-5.0f, gates[4 + q] - mf));
    ef[q] = expf(cg);
    sef += ef[q];
  }
  sef = blockSum(sef, sred);

  const float* cpr = c_prev + (size_t)row * H;
  float cpre[4]; float csum = 0.0f;
#pragma unroll
  for (int q = 0; q < 4; ++q) {
    float iv = ei[q] / (sei + 1e-6f);
    float fv = ef[q] / (sef + 1e-6f);
    float gs = iv + fv;
    iv = iv / (gs + 1e-6f);
    fv = fv / (gs + 1e-6f);
    float ctil = tanhf(gates[8 + q]);
    int p = t + 256 * q;
    cpre[q] = fv * cpr[p] + iv * ctil;
    csum += cpre[q];
  }
  float muc = blockSum(csum, sred) * invH;
  float vc = 0.0f;
#pragma unroll
  for (int q = 0; q < 4; ++q) { float d = cpre[q] - muc; vc += d * d; }
  vc = blockSum(vc, sred) * invH;
  float rsc = rsqrtf(vc + 1e-5f);

  float cv[4], tv[4]; float tsum = 0.0f;
#pragma unroll
  for (int q = 0; q < 4; ++q) {
    int p = t + 256 * q;
    cv[q] = (cpre[q] - muc) * rsc * ln_c_g[p] + ln_c_b[p];
    tv[q] = tanhf(cv[q]);
    tsum += tv[q];
  }
  float mut = blockSum(tsum, sred) * invH;
  float vt = 0.0f;
#pragma unroll
  for (int q = 0; q < 4; ++q) { float d = tv[q] - mut; vt += d * d; }
  vt = blockSum(vt, sred) * invH;
  float rst = rsqrtf(vt + 1e-5f);

  float* hrow = h_out + (size_t)row * H;
  float* crow = c_out + (size_t)row * H;
#pragma unroll
  for (int q = 0; q < 4; ++q) {
    int p = t + 256 * q;
    float co = (tv[q] - mut) * rst * ln_h_g[p] + ln_h_b[p];
    float ov = 1.0f / (1.0f + expf(-gates[12 + q]));
    hrow[p] = ov * co;
    crow[p] = cv[q];
  }
}

extern "C" void kernel_launch(void* const* d_in, const int* in_sizes, int n_in,
                              void* d_out, int out_size, void* d_ws, size_t ws_size,
                              hipStream_t stream) {
  const float* x       = (const float*)d_in[0];
  const float* h_prev  = (const float*)d_in[1];
  const float* c_prev  = (const float*)d_in[2];
  const float* Wx      = (const float*)d_in[3];
  const float* bx      = (const float*)d_in[4];
  const float* Wh      = (const float*)d_in[5];
  const float* bh      = (const float*)d_in[6];
  const float* ln_in_g = (const float*)d_in[7];
  const float* ln_in_b = (const float*)d_in[8];
  const float* ln_hu_g = (const float*)d_in[9];
  const float* ln_hu_b = (const float*)d_in[10];
  const float* ln_c_g  = (const float*)d_in[11];
  const float* ln_c_b  = (const float*)d_in[12];
  const float* ln_h_g  = (const float*)d_in[13];
  const float* ln_h_b  = (const float*)d_in[14];

  const int FH  = in_sizes[4];        // 4H (=4096)
  const int H   = FH / 4;             // 1024
  const int Din = in_sizes[3] / FH;   // 512
  const int B   = in_sizes[0] / Din;  // 8192

  float* gxbuf = (float*)d_ws;                       // [B, 4H] pre-LN x-gates
  float* ghbuf = gxbuf + (size_t)B * FH;             // [B, 4H] pre-LN h-gates
  float* h_out = (float*)d_out;                      // [B, H]
  float* c_out = h_out + (size_t)B * H;              // [B, H]

  dim3 blk(256);
  dim3 gg(B / 128, FH / 128);
  hipLaunchKernelGGL(gemm_nt_bf16x3, gg, blk, 0, stream, x, Wx, gxbuf, FH, Din);
  hipLaunchKernelGGL(gemm_nt_bf16x3, gg, blk, 0, stream, h_prev, Wh, ghbuf, FH, H);
  hipLaunchKernelGGL(slstm_pointwise, dim3(B), blk, 0, stream,
                     gxbuf, ghbuf, c_prev, bx, bh,
                     ln_in_g, ln_in_b, ln_hu_g, ln_hu_b,
                     ln_c_g, ln_c_b, ln_h_g, ln_h_b,
                     h_out, c_out, H);
}
